// VLAttLstm_4097398800758
// MI455X (gfx1250) — compile-verified, matrix operand order checked by reading, not run on hardware
//
#include <hip/hip_runtime.h>
#include <math.h>

// ---------------------------------------------------------------------------
// VLAttLstm for MI455X (gfx1250, wave32, WMMA + async global->LDS)
//   S=8192, B=1, IN=H=1024, G=4H=4096
//
// Phase 1: xg = x @ W_ih^T            -- bf16 WMMA GEMM, B tile double-buffered
//                                        into LDS via GLOBAL_LOAD_ASYNC_TO_LDS
// Phase 2: sequential LSTM scan       -- persistent kernel, W_hh slice in LDS
//                                        (132KB/WG; CDNA5 320KB WGP LDS),
//                                        grid sync via L2 atomic counter
// Phase 3: m = tanh(lstm_out @ A^T + cterm) -- same WMMA GEMM (fused epilogue)
//          scores/v-dot, global max, exp, causal cumsum (3-pass chunked scan)
// ---------------------------------------------------------------------------

typedef __attribute__((ext_vector_type(16))) __bf16 v16bf;
typedef __attribute__((ext_vector_type(8)))  __bf16 v8bf;
typedef __attribute__((ext_vector_type(8)))  float  v8f;

#define S_LEN 8192
#define H_DIM 1024
#define G_DIM 4096

// ---- CDNA5 async global->LDS copy (ASYNCcnt path) -------------------------
// One instruction moves 16B per active lane into LDS. The low 32 bits of a
// generic pointer into __shared__ are the LDS byte address (flat aperture
// truncation rule), which is exactly what VDST of the async op wants.
static __device__ __forceinline__ void async_copy_b128(const void* gsrc,
                                                       void* ldst) {
  unsigned lds = (unsigned)(size_t)ldst;
  unsigned long long ga = (unsigned long long)(size_t)gsrc;
  asm volatile("global_load_async_to_lds_b128 %0, %1, off"
               :: "v"(lds), "v"(ga)
               : "memory");
}
static __device__ __forceinline__ void wait_asynccnt0() {
  asm volatile("s_wait_asynccnt 0x0" ::: "memory");
}

// ------------------------- f32 -> bf16 conversion --------------------------
__global__ void f32_to_bf16_kernel(const float* __restrict__ in,
                                   __bf16* __restrict__ out, int n) {
  int base = blockIdx.x * blockDim.x * 4 + threadIdx.x;
#pragma unroll
  for (int k = 0; k < 4; ++k) {
    int j = base + k * blockDim.x;
    if (j < n) out[j] = (__bf16)in[j];
  }
}

// --------------------------- WMMA bf16 GEMM --------------------------------
// C[M,N] = A[M,K] * B[N,K]^T   (A,B bf16 row-major, K contiguous; C f32)
// MODE 0: plain store.  MODE 1: C = tanh(acc + cterm[n])  (attention fuse)
// Block = 256 threads = 8 waves; block tile 128(M) x 64(N); wave tile 16x64.
// B tile (64 rows x 32 k = 4KB) staged into LDS, double buffered: each wave
// issues one async b128 (512B slice), overlapping the next tile's DMA with
// this tile's 4 WMMAs; retire with s_wait_asynccnt + barrier.
#define BROW 40   // LDS row stride in elements (80B, 16B aligned, bank skew)

template <int MODE>
__global__ __launch_bounds__(256) void wmma_gemm_kernel(
    const __bf16* __restrict__ A, const __bf16* __restrict__ B,
    float* __restrict__ C, const float* __restrict__ cterm,
    int M, int N, int K) {
  __shared__ __bf16 Btile[2][64 * BROW];   // 2 x 5KB

  const int tid = threadIdx.x;
  const int wave = tid >> 5;
  const int lane = tid & 31;
  const int m0 = (blockIdx.y * 8 + wave) * 16;
  const int n0 = blockIdx.x * 64;
  const int arow = m0 + (lane & 15);
  const int kb = (lane < 16) ? 0 : 8;   // ISA 16-bit A/B lane K-offset

  // Per-thread staging slice: 256 threads x 16B cover the 64x32 bf16 tile.
  const int srow = tid >> 2;            // 0..63  (row of B tile)
  const int scol = (tid & 3) * 8;       // 0,8,16,24 (elements within k-step)
  const __bf16* bsrc = B + (size_t)(n0 + srow) * K + scol;
  __bf16* bdst0 = &Btile[0][srow * BROW + scol];
  __bf16* bdst1 = &Btile[1][srow * BROW + scol];

  // Prologue: stage k0 = 0 into buffer 0.
  async_copy_b128(bsrc, bdst0);
  wait_asynccnt0();
  __syncthreads();

  v8f acc[4];
#pragma unroll
  for (int nt = 0; nt < 4; ++nt) {
    v8f z = {0.f, 0.f, 0.f, 0.f, 0.f, 0.f, 0.f, 0.f};
    acc[nt] = z;
  }

  int p = 0;
  for (int k0 = 0; k0 < K; k0 += 32) {
    // Kick off DMA of the next B tile into the other buffer.
    if (k0 + 32 < K)
      async_copy_b128(bsrc + (k0 + 32), p ? bdst0 : bdst1);

    // A fragment from global (per-wave rows, streams through L2).
    const __bf16* ap = A + (size_t)arow * K + (k0 + kb);
    v8bf alo = *(const v8bf*)ap;
    v8bf ahi = *(const v8bf*)(ap + 16);
    v16bf a;
#pragma unroll
    for (int i = 0; i < 8; ++i) { a[i] = alo[i]; a[i + 8] = ahi[i]; }

    const __bf16* btb = &Btile[p][0];
#pragma unroll
    for (int nt = 0; nt < 4; ++nt) {
      const __bf16* bp = btb + (nt * 16 + (lane & 15)) * BROW + kb;
      v8bf blo = *(const v8bf*)bp;
      v8bf bhi = *(const v8bf*)(bp + 16);
      v16bf b;
#pragma unroll
      for (int i = 0; i < 8; ++i) { b[i] = blo[i]; b[i + 8] = bhi[i]; }
      acc[nt] = __builtin_amdgcn_wmma_f32_16x16x32_bf16(
          false, a, false, b, (short)0, acc[nt], false, false);
    }

    // Retire this step's DMA (overlapped with the WMMAs above), publish tile.
    wait_asynccnt0();
    __syncthreads();
    p ^= 1;
  }

  // C/D layout: lanes 0-15 -> M=m0..m0+7 (vgpr 0..7), lanes 16-31 -> M=m0+8..15
  const int mbase = m0 + ((lane >> 4) << 3);
#pragma unroll
  for (int nt = 0; nt < 4; ++nt) {
    const int n = n0 + nt * 16 + (lane & 15);
    float ct = (MODE == 1) ? cterm[n] : 0.f;
#pragma unroll
    for (int v = 0; v < 8; ++v) {
      float val = acc[nt][v];
      if (MODE == 1) val = tanhf(val + ct);
      C[(size_t)(mbase + v) * N + n] = val;
    }
  }
}

// --------------------------- persistent LSTM scan --------------------------
#define SCAN_WGS 128
#define SCAN_THREADS 256
#define JPW 8     // hidden elements owned per WG
#define ROWS 32   // 4 gates * JPW rows of W_hh per WG
#define WPAD 1032 // row pad -> 8-bank skew, kills 4-way LDS conflicts

__global__ __launch_bounds__(SCAN_THREADS) void lstm_scan_kernel(
    const float* __restrict__ xg, const float* __restrict__ W_hh,
    const float* __restrict__ b_ih, const float* __restrict__ b_hh,
    float* __restrict__ lstm_out, __bf16* __restrict__ lstm_bf,
    float* __restrict__ h_glob, int* __restrict__ counter) {
  __shared__ float Wsl[ROWS][WPAD];   // 132 KB: CDNA5 320KB WGP LDS
  __shared__ float h_sh[H_DIM];
  __shared__ float gacc[ROWS];
  __shared__ float gbias[ROWS];
  __shared__ float c_sh[JPW];

  const int t = threadIdx.x;
  const int j0 = blockIdx.x * JPW;

  // Cache this WG's 32 W_hh rows in LDS once (never re-read from memory).
  for (int rr = 0; rr < ROWS; ++rr) {
    int grow = (rr >> 3) * H_DIM + j0 + (rr & 7);
    const float* src = W_hh + (size_t)grow * H_DIM;
    for (int k = t; k < H_DIM; k += SCAN_THREADS) Wsl[rr][k] = src[k];
  }
  if (t < ROWS) {
    int grow = (t >> 3) * H_DIM + j0 + (t & 7);
    gbias[t] = b_ih[grow] + b_hh[grow];
  }
  if (t < JPW) c_sh[t] = 0.f;
  for (int k = t; k < H_DIM; k += SCAN_THREADS) h_sh[k] = 0.f;
  __syncthreads();

  const int row = t >> 3;   // 0..31
  const int seg = t & 7;    // 8 lanes cooperate per row (128 elems each)

  for (int s = 0; s < S_LEN; ++s) {
    const float* wrow = &Wsl[row][seg * 128];
    const float* hseg = &h_sh[seg * 128];
    float p0 = 0.f, p1 = 0.f, p2 = 0.f, p3 = 0.f;
#pragma unroll 8
    for (int k = 0; k < 128; k += 4) {
      p0 += wrow[k + 0] * hseg[k + 0];
      p1 += wrow[k + 1] * hseg[k + 1];
      p2 += wrow[k + 2] * hseg[k + 2];
      p3 += wrow[k + 3] * hseg[k + 3];
    }
    float p = (p0 + p1) + (p2 + p3);
    // reduce across the 8 seg lanes (contiguous lanes in one wave32)
    p += __shfl_xor(p, 1, 32);
    p += __shfl_xor(p, 2, 32);
    p += __shfl_xor(p, 4, 32);
    if (seg == 0) {
      int grow = (row >> 3) * H_DIM + j0 + (row & 7);
      gacc[row] = p + gbias[row] + xg[(size_t)s * G_DIM + grow];
    }
    __syncthreads();

    if (t < JPW) {
      float iv = gacc[t], fv = gacc[8 + t], gv = gacc[16 + t], ov = gacc[24 + t];
      iv = 1.f / (1.f + __expf(-iv));
      fv = 1.f / (1.f + __expf(-fv));
      gv = tanhf(gv);
      ov = 1.f / (1.f + __expf(-ov));
      float c = fv * c_sh[t] + iv * gv;
      c_sh[t] = c;
      float h = ov * tanhf(c);
      int j = j0 + t;
      h_glob[j] = h;
      lstm_out[(size_t)s * H_DIM + j] = h;
      lstm_bf[(size_t)s * H_DIM + j] = (__bf16)h;
    }

    // ---- grid-wide barrier (monotonic counter; WGs must be co-resident) ----
    __threadfence();
    __syncthreads();
    if (t == 0) {
      __hip_atomic_fetch_add(counter, 1, __ATOMIC_RELEASE,
                             __HIP_MEMORY_SCOPE_AGENT);
      const int target = (s + 1) * SCAN_WGS;
      while (__hip_atomic_load(counter, __ATOMIC_ACQUIRE,
                               __HIP_MEMORY_SCOPE_AGENT) < target) {
        __builtin_amdgcn_s_sleep(1);
      }
    }
    __syncthreads();
    for (int k = t; k < H_DIM; k += SCAN_THREADS) h_sh[k] = h_glob[k];
    __syncthreads();
  }
}

// -------- cterm[n] = att_cw_w[n,:]·h_n + cw_b + hw_b + att_bias ------------
__global__ void cterm_kernel(const float* __restrict__ h_n,
                             const float* __restrict__ cw_w,
                             const float* __restrict__ cw_b,
                             const float* __restrict__ hw_b,
                             const float* __restrict__ att_bias,
                             float* __restrict__ cterm) {
  int n = blockIdx.x * blockDim.x + threadIdx.x;
  if (n >= H_DIM) return;
  const float* row = cw_w + (size_t)n * H_DIM;
  float acc = 0.f;
  for (int k = 0; k < H_DIM; ++k) acc += row[k] * h_n[k];
  cterm[n] = acc + cw_b[n] + hw_b[n] + att_bias[n];
}

// -------------- scores[s] = m[s,:]·v + vb  (one wave per s) ----------------
__global__ void scores_kernel(const float* __restrict__ m_buf,
                              const float* __restrict__ v2s_w,
                              const float* __restrict__ v2s_b,
                              float* __restrict__ scores) {
  int wave = threadIdx.x >> 5, lane = threadIdx.x & 31;
  int s = blockIdx.x * 8 + wave;
  const float* row = m_buf + (size_t)s * H_DIM;
  float acc = 0.f;
  for (int k = lane; k < H_DIM; k += 32) acc += row[k] * v2s_w[k];
  for (int off = 16; off; off >>= 1) acc += __shfl_xor(acc, off, 32);
  if (lane == 0) scores[s] = acc + v2s_b[0];
}

// ----- single block: global max, e[s]=exp(score-max), den_cum = cumsum(e) --
__global__ __launch_bounds__(1024) void attn_scalar_kernel(
    const float* __restrict__ scores, float* __restrict__ e,
    float* __restrict__ den_cum) {
  __shared__ float red[32];
  __shared__ float chunk[1024];
  __shared__ float carry_sh;
  int t = threadIdx.x;

  float mx = -INFINITY;
  for (int s = t; s < S_LEN; s += 1024) mx = fmaxf(mx, scores[s]);
  for (int off = 16; off; off >>= 1) mx = fmaxf(mx, __shfl_xor(mx, off, 32));
  if ((t & 31) == 0) red[t >> 5] = mx;
  __syncthreads();
  if (t < 32) {
    float v = red[t];
    for (int off = 16; off; off >>= 1) v = fmaxf(v, __shfl_xor(v, off, 32));
    if (t == 0) red[0] = v;
  }
  __syncthreads();
  mx = red[0];
  if (t == 0) carry_sh = 0.f;
  __syncthreads();

  for (int base = 0; base < S_LEN; base += 1024) {
    float ev = __expf(scores[base + t] - mx);
    e[base + t] = ev;
    chunk[t] = ev;
    __syncthreads();
    for (int off = 1; off < 1024; off <<= 1) {   // Hillis-Steele inclusive scan
      float add = (t >= off) ? chunk[t - off] : 0.f;
      __syncthreads();
      chunk[t] += add;
      __syncthreads();
    }
    float total = chunk[1023];
    den_cum[base + t] = chunk[t] + carry_sh;
    __syncthreads();
    if (t == 0) carry_sh += total;
    __syncthreads();
  }
}

// ------------------ chunked causal cumsum over channels --------------------
#define NCHUNK 64
#define SCHUNK (S_LEN / NCHUNK)

__global__ __launch_bounds__(256) void chunk_sums_kernel(
    const float* __restrict__ e, const float* __restrict__ lstm_out,
    float* __restrict__ chunksum) {
  int b = blockIdx.x, t = threadIdx.x;
  float acc[4] = {0.f, 0.f, 0.f, 0.f};
  int s0 = b * SCHUNK;
  for (int s = s0; s < s0 + SCHUNK; ++s) {
    float ev = e[s];
    const float* row = lstm_out + (size_t)s * H_DIM;
#pragma unroll
    for (int q = 0; q < 4; ++q) acc[q] += ev * row[t + q * 256];
  }
#pragma unroll
  for (int q = 0; q < 4; ++q) chunksum[(size_t)b * H_DIM + t + q * 256] = acc[q];
}

__global__ void chunk_prefix_kernel(float* __restrict__ chunksum) {
  int t = threadIdx.x;
#pragma unroll
  for (int q = 0; q < 4; ++q) {
    int h = t + q * 256;
    float run = 0.f;
    for (int b = 0; b < NCHUNK; ++b) {
      float v = chunksum[(size_t)b * H_DIM + h];
      chunksum[(size_t)b * H_DIM + h] = run;  // exclusive prefix in place
      run += v;
    }
  }
}

__global__ __launch_bounds__(256) void attn_out_kernel(
    const float* __restrict__ e, const float* __restrict__ lstm_out,
    const float* __restrict__ chunksum, const float* __restrict__ den_cum,
    float* __restrict__ out) {
  int b = blockIdx.x, t = threadIdx.x;
  float run[4];
#pragma unroll
  for (int q = 0; q < 4; ++q) run[q] = chunksum[(size_t)b * H_DIM + t + q * 256];
  int s0 = b * SCHUNK;
  for (int s = s0; s < s0 + SCHUNK; ++s) {
    float ev = e[s];
    float inv = 1.f / den_cum[s];
    const float* row = lstm_out + (size_t)s * H_DIM;
    float* orow = out + (size_t)s * H_DIM;
#pragma unroll
    for (int q = 0; q < 4; ++q) {
      run[q] += ev * row[t + q * 256];
      orow[t + q * 256] = run[q] * inv;
    }
  }
}

__global__ void init_kernel(int* counter) { *counter = 0; }

// ------------------------------- launcher ----------------------------------
extern "C" void kernel_launch(void* const* d_in, const int* in_sizes, int n_in,
                              void* d_out, int out_size, void* d_ws,
                              size_t ws_size, hipStream_t stream) {
  (void)in_sizes; (void)n_in; (void)out_size; (void)ws_size;
  const float* x        = (const float*)d_in[0];
  const float* W_ih     = (const float*)d_in[1];
  const float* W_hh     = (const float*)d_in[2];
  const float* b_ih     = (const float*)d_in[3];
  const float* b_hh     = (const float*)d_in[4];
  const float* att_hw_w = (const float*)d_in[5];
  const float* att_hw_b = (const float*)d_in[6];
  const float* att_cw_w = (const float*)d_in[7];
  const float* att_cw_b = (const float*)d_in[8];
  const float* att_bias = (const float*)d_in[9];
  const float* att_v2s_w = (const float*)d_in[10];
  const float* att_v2s_b = (const float*)d_in[11];
  float* out = (float*)d_out;

  // ---- workspace carve-out (all offsets 256B aligned) ----
  char* ws = (char*)d_ws;
  size_t off = 0;
  auto carve = [&](size_t bytes) {
    size_t p = off;
    off += (bytes + 255) & ~(size_t)255;
    return p;
  };
  size_t o_xg      = carve((size_t)S_LEN * G_DIM * 4);   // 128 MB (reused as m_buf)
  size_t o_xbf     = carve((size_t)S_LEN * H_DIM * 2);
  size_t o_wihbf   = carve((size_t)G_DIM * H_DIM * 2);
  size_t o_attwbf  = carve((size_t)H_DIM * H_DIM * 2);
  size_t o_lstm    = carve((size_t)S_LEN * H_DIM * 4);
  size_t o_lstmbf  = carve((size_t)S_LEN * H_DIM * 2);
  size_t o_hglob   = carve((size_t)H_DIM * 4);
  size_t o_cterm   = carve((size_t)H_DIM * 4);
  size_t o_scores  = carve((size_t)S_LEN * 4);
  size_t o_e       = carve((size_t)S_LEN * 4);
  size_t o_dencum  = carve((size_t)S_LEN * 4);
  size_t o_csum    = carve((size_t)NCHUNK * H_DIM * 4);
  size_t o_counter = carve(256);

  float*  xg      = (float*)(ws + o_xg);
  __bf16* x_bf    = (__bf16*)(ws + o_xbf);
  __bf16* wih_bf  = (__bf16*)(ws + o_wihbf);
  __bf16* attw_bf = (__bf16*)(ws + o_attwbf);
  float*  lstm_f  = (float*)(ws + o_lstm);
  __bf16* lstm_bf = (__bf16*)(ws + o_lstmbf);
  float*  h_glob  = (float*)(ws + o_hglob);
  float*  cterm   = (float*)(ws + o_cterm);
  float*  m_buf   = xg;                      // xg dead after the scan; alias
  float*  scores  = (float*)(ws + o_scores);
  float*  e_buf   = (float*)(ws + o_e);
  float*  den_cum = (float*)(ws + o_dencum);
  float*  csum    = (float*)(ws + o_csum);
  int*    counter = (int*)(ws + o_counter);

  init_kernel<<<1, 1, 0, stream>>>(counter);

  // f32 -> bf16 conversions
  {
    int n = S_LEN * H_DIM;
    f32_to_bf16_kernel<<<(n + 1023) / 1024, 256, 0, stream>>>(x, x_bf, n);
    n = G_DIM * H_DIM;
    f32_to_bf16_kernel<<<(n + 1023) / 1024, 256, 0, stream>>>(W_ih, wih_bf, n);
    n = H_DIM * H_DIM;
    f32_to_bf16_kernel<<<(n + 1023) / 1024, 256, 0, stream>>>(att_hw_w, attw_bf, n);
  }

  // Phase 1: xg = x @ W_ih^T   (WMMA + async B staging)
  wmma_gemm_kernel<0><<<dim3(G_DIM / 64, S_LEN / 128), 256, 0, stream>>>(
      x_bf, wih_bf, xg, nullptr, S_LEN, G_DIM, H_DIM);

  // Phase 2: sequential LSTM scan (persistent, grid-synced)
  lstm_scan_kernel<<<SCAN_WGS, SCAN_THREADS, 0, stream>>>(
      xg, W_hh, b_ih, b_hh, lstm_f, lstm_bf, h_glob, counter);

  // Phase 3: attention
  cterm_kernel<<<H_DIM / 256, 256, 0, stream>>>(h_glob, att_cw_w, att_cw_b,
                                                att_hw_b, att_bias, cterm);
  wmma_gemm_kernel<1><<<dim3(H_DIM / 64, S_LEN / 128), 256, 0, stream>>>(
      lstm_bf, attw_bf, m_buf, cterm, S_LEN, H_DIM, H_DIM);
  scores_kernel<<<S_LEN / 8, 256, 0, stream>>>(m_buf, att_v2s_w, att_v2s_b,
                                               scores);
  attn_scalar_kernel<<<1, 1024, 0, stream>>>(scores, e_buf, den_cum);
  chunk_sums_kernel<<<NCHUNK, 256, 0, stream>>>(e_buf, lstm_f, csum);
  chunk_prefix_kernel<<<1, 256, 0, stream>>>(csum);
  attn_out_kernel<<<NCHUNK, 256, 0, stream>>>(e_buf, lstm_f, csum, den_cum,
                                              out);
}
